// SNRGuidedCrossViewFusion_22514218565904
// MI455X (gfx1250) — matrix operand order repaired and checked by reading, not compiled, not run on hardware
//
#include <hip/hip_runtime.h>
#include <hip/hip_bf16.h>
#include <math.h>

#define B_   4
#define N_   6
#define C_   128
#define P_   16384
#define MID_ 32
#define LDST 136   // padded bf16 stride per pixel row in LDS (272B -> conflict-free)

typedef __attribute__((ext_vector_type(16))) __bf16 v16bf;
typedef __attribute__((ext_vector_type(8)))  float  v8f;
typedef unsigned int v4u __attribute__((ext_vector_type(4)));
typedef int          v4i __attribute__((ext_vector_type(4)));
typedef int          v8i __attribute__((ext_vector_type(8)));

#define BPC      (B_ * P_ * C_)        /* 8,388,608 */
#define WS_REL   0
#define WS_GATE  32
#define WS_GQ    1024
#define WS_FC    (1024 + BPC)
#define WS_MID_F (1024 + 2 * BPC)      /* float offset; region reinterpreted as bf16 */

#if __has_builtin(__builtin_amdgcn_tensor_load_to_lds) && \
    __has_builtin(__builtin_amdgcn_tensor_store_from_lds) && \
    __has_builtin(__builtin_amdgcn_s_wait_tensorcnt)
#define HAVE_TDM 1
#else
#define HAVE_TDM 0
#endif

__device__ inline v8f splat8(float x) {
  v8f v;
#pragma unroll
  for (int i = 0; i < 8; ++i) v[i] = x;
  return v;
}

__device__ inline v8f wmma_bf16(v16bf a, v16bf b, v8f c) {
  return __builtin_amdgcn_wmma_f32_16x16x32_bf16(false, a, false, b, (short)0, c,
                                                 false, false);
}

#if HAVE_TDM
// Build a D# for a 2D tile (tile0 contiguous elems, tile1 lines of stride0) and
// issue TENSOR_LOAD_TO_LDS.  dsz: 0=1B,1=2B,2=4B.  (ISA 08_async_tensor §8)
__device__ inline void tdm_load_2d(const void* g, void* l, int tile0, int tile1,
                                   int tdim0, int tdim1, int stride0, int dsz) {
  unsigned long long ga = (unsigned long long)(size_t)g;
  v4u g0;
  g0[0] = 1u;                                   // count=1 (valid descriptor)
  g0[1] = (unsigned int)(size_t)l;              // lds_addr (low 32 of flat = offset)
  g0[2] = (unsigned int)ga;                     // global_addr[31:0]
  g0[3] = ((unsigned int)(ga >> 32) & 0x01FFFFFFu) | 0x80000000u; // addr hi | type=2
  v8i g1;
  g1[0] = dsz << 16;                            // wg_mask=0, data_size
  g1[1] = (int)((tdim0 & 0xFFFF) << 16);        // tensor_dim0[15:0]
  g1[2] = (int)(((unsigned)tdim0 >> 16) | (((unsigned)tdim1 & 0xFFFF) << 16));
  g1[3] = (int)((((unsigned)tdim1 >> 16) & 0xFFFF) | ((unsigned)tile0 << 16));
  g1[4] = tile1 & 0xFFFF;                       // tile_dim1, tile_dim2=0
  g1[5] = stride0;                              // tensor_dim0_stride[31:0]
  g1[6] = 0;
  g1[7] = 0;
  v4i z4 = {0, 0, 0, 0};
  v8i z8 = {0, 0, 0, 0, 0, 0, 0, 0};
  __builtin_amdgcn_tensor_load_to_lds(g0, g1, z4, z4, z8, 0);
}

__device__ inline void tdm_store_1d(void* g, const void* l, int elems, int dsz) {
  unsigned long long ga = (unsigned long long)(size_t)g;
  v4u g0;
  g0[0] = 1u;
  g0[1] = (unsigned int)(size_t)l;
  g0[2] = (unsigned int)ga;
  g0[3] = ((unsigned int)(ga >> 32) & 0x01FFFFFFu) | 0x80000000u;
  v8i g1;
  g1[0] = dsz << 16;
  g1[1] = (int)((elems & 0xFFFF) << 16);
  g1[2] = (int)((unsigned)elems >> 16);
  g1[3] = (int)((unsigned)(elems & 0xFFFF) << 16); // tile_dim0 (<=65535 for our uses)
  g1[4] = 0;
  g1[5] = elems;
  g1[6] = 0;
  g1[7] = 0;
  v4i z4 = {0, 0, 0, 0};
  v8i z8 = {0, 0, 0, 0, 0, 0, 0, 0};
  __builtin_amdgcn_tensor_store_from_lds(g0, g1, z4, z4, z8, 0);
}
#endif

// A fragment (16 pixels x 32 ch) from LDS, pixel-major bf16 with stride LDST.
// ISA 16-bit A layout: lane m=lane&15 holds row m; k-halves split by lane>>4.
__device__ inline v16bf a_from_lds(const __bf16* buf, int lane, int ks) {
  const int m  = lane & 15;
  const int kb = ks * 32 + (lane >> 4) * 8;
  const __bf16* r = buf + m * LDST + kb;
  v16bf a;
#pragma unroll
  for (int j = 0; j < 8; ++j) { a[j] = r[j]; a[8 + j] = r[16 + j]; }
  return a;
}

// B fragment (32 x 16) from a row-major f32 weight matrix W[O][ldw]:
// lane col o = nt*16 + (lane&15); rows K = ks*32 + (lane>>4)*8 + j (j<8), +16 for j>=8.
__device__ inline v16bf b_from_w(const float* W, int ldw, int nt, int lane, int ks) {
  const int o = nt * 16 + (lane & 15);
  const int k = ks * 32 + (lane >> 4) * 8;
  const float* r = W + o * ldw + k;
  v16bf b;
#pragma unroll
  for (int j = 0; j < 16; ++j) b[j] = (__bf16)r[j + (j >> 3) * 8];
  return b;
}

// ---------------- K1: tiny MLPs (rel_bias per (b,n), SNR gate per (b,c)) ---
__global__ void k_prep(const float* __restrict__ snr, const float* __restrict__ vr,
                       const float* __restrict__ Wg, const float* __restrict__ bg,
                       const float* __restrict__ Wr1, const float* __restrict__ br1,
                       const float* __restrict__ Wr2, const float* __restrict__ br2,
                       float* __restrict__ ws) {
  int idx = blockIdx.x * blockDim.x + threadIdx.x;
  if (idx < B_ * N_) {
    float acc = br2[0];
#pragma unroll
    for (int h = 0; h < MID_; ++h) {
      float t = br1[h];
      for (int j = 0; j < 3; ++j) t += vr[idx * 3 + j] * Wr1[h * 3 + j];
      acc += fmaxf(t, 0.f) * Wr2[h];
    }
    ws[WS_REL + idx] = acc;
  } else if (idx < B_ * N_ + B_ * C_) {
    int i = idx - B_ * N_;
    int b = i >> 7, c = i & 127;
    float sn = snr[b] * (1.f / 20.f);
    sn = fminf(fmaxf(sn, -1.5f), 1.5f);
    float z = sn * Wg[c] + bg[c];
    ws[WS_GATE + i] = 1.f / (1.f + __expf(-z));
  }
}

// ---------------- K2: cross-view attention + conf fusion (per 16-px tile) ---
__global__ void __launch_bounds__(32)
k_attn(const float* __restrict__ bev,
       const float* __restrict__ Wq,  const float* __restrict__ bq,
       const float* __restrict__ Wk,  const float* __restrict__ bk,
       const float* __restrict__ Wv,  const float* __restrict__ bv,
       const float* __restrict__ Wo1, const float* __restrict__ bo1,
       const float* __restrict__ Wc1, const float* __restrict__ bc1,
       const float* __restrict__ Wc2, const float* __restrict__ bc2,
       float* __restrict__ ws,
       float* __restrict__ out_attn, float* __restrict__ out_conf) {
  __shared__ __bf16 xs[N_][16 * LDST];   // staged views (bf16, pixel-major)
  __shared__ __bf16 fb[16 * LDST];       // gq, then fused (reused)
  __shared__ float  stage[2][16 * C_];   // TDM f32 landing buffers (8KB each)
  __shared__ float  sc[N_][4][16];       // scores  [view][head][pixel]
  __shared__ float  aw[N_][4][16];       // attn    [view][head][pixel]
  __shared__ float  cl[N_][16];          // conf logits
  __shared__ float  cw[N_][16];          // conf weights

  const int lane = threadIdx.x;
  const int tile = blockIdx.x;
  const int b    = tile >> 10;
  const int p0   = (tile & 1023) * 16;
  const int m    = lane & 15;
  const int kh   = lane >> 4;

  float*  gq_ws = ws + WS_GQ;
  float*  fc_ws = ws + WS_FC;
  __bf16* mid   = (__bf16*)(ws + WS_MID_F);

  __builtin_prefetch(Wq, 0, 1);
  __builtin_prefetch(Wk, 0, 1);
  __builtin_prefetch(Wv, 0, 1);
  __builtin_prefetch(Wo1, 0, 1);
  __builtin_prefetch(Wc1, 0, 1);

  // ---- stage views into LDS as bf16 (pixel-major) ----
#if HAVE_TDM
  // Double-buffered TDM: 2D tile = 16 contiguous pixels x 128 channel lines
  // (stride 16384 f32).  One tensor op replaces 2048 scalar global loads (64 per lane).
  tdm_load_2d(bev + ((size_t)(b * N_ + 0) * C_) * P_ + p0, stage[0],
              16, C_, P_, C_, P_, 2);
  for (int n = 0; n + 1 < N_; ++n) {
    // TDM is not ordered against this wave's DS reads of the buffer it is
    // about to overwrite -> drain DS first.
    asm volatile("s_wait_dscnt 0x0" ::: "memory");
    tdm_load_2d(bev + ((size_t)(b * N_ + n + 1) * C_) * P_ + p0,
                stage[(n + 1) & 1], 16, C_, P_, C_, P_, 2);
    __builtin_amdgcn_s_wait_tensorcnt(1);   // oldest load done (in-order)
    __syncthreads();
    const float* sg = stage[n & 1];
    for (int e = lane; e < 16 * C_; e += 32) {        // sg[c*16+px]
      xs[n][(e & 15) * LDST + (e >> 4)] = (__bf16)sg[e];
    }
    __syncthreads();
  }
  __builtin_amdgcn_s_wait_tensorcnt(0);
  __syncthreads();
  {
    const float* sg = stage[(N_ - 1) & 1];
    for (int e = lane; e < 16 * C_; e += 32) {
      xs[N_ - 1][(e & 15) * LDST + (e >> 4)] = (__bf16)sg[e];
    }
  }
#else
  for (int n = 0; n < N_; ++n) {
    const float* src = bev + ((size_t)(b * N_ + n) * C_) * P_ + p0;
    for (int e = lane; e < 16 * C_; e += 32) {
      int c = e >> 4, px = e & 15;
      xs[n][px * LDST + c] = (__bf16)src[(size_t)c * P_ + px];
    }
  }
#endif
  __syncthreads();

  // gq = mean over views (bf16 in LDS for q; f32 to ws for the residual)
  for (int e = lane; e < 16 * C_; e += 32) {
    int px = e >> 7, c = e & 127;
    float s = 0.f;
    for (int n = 0; n < N_; ++n) s += (float)xs[n][px * LDST + c];
    s *= (1.f / 6.f);
    fb[px * LDST + c] = (__bf16)s;
    gq_ws[((size_t)(b * P_) + p0 + px) * C_ + c] = s;
  }
  __syncthreads();

  // q = gq @ Wq^T + bq    (8 n-tiles x 4 k-steps of 16x16x32 WMMA)
  v16bf aq[4];
#pragma unroll
  for (int ks = 0; ks < 4; ++ks) aq[ks] = a_from_lds(fb, lane, ks);
  v8f qf[8];
#pragma unroll
  for (int nt = 0; nt < 8; ++nt) {
    v8f acc = splat8(bq[nt * 16 + m]);
#pragma unroll
    for (int ks = 0; ks < 4; ++ks)
      acc = wmma_bf16(aq[ks], b_from_w(Wq, C_, nt, lane, ks), acc);
    qf[nt] = acc;
  }

  const float rscale = 0.17677669529663687f;  // 1/sqrt(HD=32)

  // per view: k GEMM, per-head dot(q,k) via 16-lane xor reductions
  for (int n = 0; n < N_; ++n) {
    v16bf ax[4];
#pragma unroll
    for (int ks = 0; ks < 4; ++ks) ax[ks] = a_from_lds(xs[n], lane, ks);
#pragma unroll
    for (int h = 0; h < 4; ++h) {
      v8f k0 = splat8(bk[(2 * h) * 16 + m]);
      v8f k1 = splat8(bk[(2 * h + 1) * 16 + m]);
#pragma unroll
      for (int ks = 0; ks < 4; ++ks) {
        k0 = wmma_bf16(ax[ks], b_from_w(Wk, C_, 2 * h, lane, ks), k0);
        k1 = wmma_bf16(ax[ks], b_from_w(Wk, C_, 2 * h + 1, lane, ks), k1);
      }
      v8f pr = qf[2 * h] * k0 + qf[2 * h + 1] * k1;
#pragma unroll
      for (int j = 0; j < 8; ++j) {
        float s = pr[j];
        s += __shfl_xor(s, 1, 32);
        s += __shfl_xor(s, 2, 32);
        s += __shfl_xor(s, 4, 32);
        s += __shfl_xor(s, 8, 32);
        if (m == 0) sc[n][h][j + 8 * kh] = s * rscale;
      }
    }
  }
  __syncthreads();

  // softmax over views per (head, pixel); attn_map = mean over heads
  for (int idx = lane; idx < 64; idx += 32) {
    int h = idx >> 4, mm = idx & 15;
    float mx = -1e30f;
    for (int n = 0; n < N_; ++n) mx = fmaxf(mx, sc[n][h][mm]);
    float se = 0.f, ex[N_];
    for (int n = 0; n < N_; ++n) { ex[n] = __expf(sc[n][h][mm] - mx); se += ex[n]; }
    float inv = 1.f / se;
    for (int n = 0; n < N_; ++n) aw[n][h][mm] = ex[n] * inv;
  }
  __syncthreads();
  for (int idx = lane; idx < N_ * 16; idx += 32) {
    int n = idx >> 4, mm = idx & 15;
    float s = (aw[n][0][mm] + aw[n][1][mm] + aw[n][2][mm] + aw[n][3][mm]) * 0.25f;
    out_attn[(size_t)(b * N_ + n) * P_ + p0 + mm] = s;
  }
  __syncthreads();

  // fused = sum_n attn * (x_n @ Wv^T + bv)
  v8f ff[8];
#pragma unroll
  for (int nt = 0; nt < 8; ++nt) ff[nt] = splat8(0.f);
  for (int n = 0; n < N_; ++n) {
    v16bf ax[4];
#pragma unroll
    for (int ks = 0; ks < 4; ++ks) ax[ks] = a_from_lds(xs[n], lane, ks);
#pragma unroll
    for (int nt = 0; nt < 8; ++nt) {
      v8f acc = splat8(bv[nt * 16 + m]);
#pragma unroll
      for (int ks = 0; ks < 4; ++ks)
        acc = wmma_bf16(ax[ks], b_from_w(Wv, C_, nt, lane, ks), acc);
      int h = nt >> 1;
#pragma unroll
      for (int j = 0; j < 8; ++j)
        ff[nt][j] += aw[n][h][j + 8 * kh] * acc[j];
    }
  }

  // confidence: c2( relu( c1(x_n) ) ) + bc2 + rel_bias, softmax over views
  const float w2a = Wc2[m], w2b = Wc2[16 + m];
  for (int n = 0; n < N_; ++n) {
    v16bf ax[4];
#pragma unroll
    for (int ks = 0; ks < 4; ++ks) ax[ks] = a_from_lds(xs[n], lane, ks);
    v8f c0 = splat8(bc1[m]);
    v8f c1 = splat8(bc1[16 + m]);
#pragma unroll
    for (int ks = 0; ks < 4; ++ks) {
      c0 = wmma_bf16(ax[ks], b_from_w(Wc1, C_, 0, lane, ks), c0);
      c1 = wmma_bf16(ax[ks], b_from_w(Wc1, C_, 1, lane, ks), c1);
    }
#pragma unroll
    for (int j = 0; j < 8; ++j) {
      float s = fmaxf(c0[j], 0.f) * w2a + fmaxf(c1[j], 0.f) * w2b;
      s += __shfl_xor(s, 1, 32);
      s += __shfl_xor(s, 2, 32);
      s += __shfl_xor(s, 4, 32);
      s += __shfl_xor(s, 8, 32);
      if (m == 0) cl[n][j + 8 * kh] = s;
    }
  }
  __syncthreads();
  if (lane < 16) {
    float lg[N_], mx = -1e30f;
    for (int n = 0; n < N_; ++n) {
      lg[n] = cl[n][lane] + bc2[0] + ws[WS_REL + b * N_ + n];
      mx = fmaxf(mx, lg[n]);
    }
    float se = 0.f, ex[N_];
    for (int n = 0; n < N_; ++n) { ex[n] = __expf(lg[n] - mx); se += ex[n]; }
    float inv = 1.f / se;
    for (int n = 0; n < N_; ++n) {
      float w = ex[n] * inv;
      cw[n][lane] = w;
      out_conf[(size_t)(b * N_ + n) * P_ + p0 + lane] = w;
    }
  }
  __syncthreads();
  // fused_conf = sum_n cw * x_n   (f32, pixel-major to ws)
  for (int e = lane; e < 16 * C_; e += 32) {
    int px = e >> 7, c = e & 127;
    float s = 0.f;
    for (int n = 0; n < N_; ++n) s += cw[n][px] * (float)xs[n][px * LDST + c];
    fc_ws[((size_t)(b * P_) + p0 + px) * C_ + c] = s;
  }
  __syncthreads();

  // mid = relu( fused @ Wo1^T + bo1 ) -> ws as bf16 (B,P,C)
#pragma unroll
  for (int nt = 0; nt < 8; ++nt)
#pragma unroll
    for (int j = 0; j < 8; ++j)
      fb[(j + 8 * kh) * LDST + nt * 16 + m] = (__bf16)ff[nt][j];
  __syncthreads();
  v16bf af[4];
#pragma unroll
  for (int ks = 0; ks < 4; ++ks) af[ks] = a_from_lds(fb, lane, ks);
  v8f of[8];
#pragma unroll
  for (int nt = 0; nt < 8; ++nt) {
    v8f acc = splat8(bo1[nt * 16 + m]);
#pragma unroll
    for (int ks = 0; ks < 4; ++ks)
      acc = wmma_bf16(af[ks], b_from_w(Wo1, C_, nt, lane, ks), acc);
    of[nt] = acc;
  }
  __syncthreads();
  // compact bf16 tile (reuse stage[0] storage), then one tensor store
  __bf16* midc = (__bf16*)&stage[0][0];
#pragma unroll
  for (int nt = 0; nt < 8; ++nt)
#pragma unroll
    for (int j = 0; j < 8; ++j)
      midc[(j + 8 * kh) * C_ + nt * 16 + m] = (__bf16)fmaxf(of[nt][j], 0.f);
  __syncthreads();
#if HAVE_TDM
  asm volatile("s_wait_dscnt 0x0" ::: "memory");   // midc writes visible to TDM
  tdm_store_1d(mid + ((size_t)(b * P_) + p0) * C_, midc, 16 * C_ * 2, 0); // bytes
#else
  for (int g = lane; g < 256; g += 32) {       // 8-half vector copies
    int px = g >> 4, c0i = (g & 15) * 8;
    const __bf16* s = &midc[px * C_ + c0i];
    __bf16* d = &mid[((size_t)(b * P_) + p0 + px) * C_ + c0i];
#pragma unroll
    for (int j = 0; j < 8; ++j) d[j] = s[j];
  }
#endif
}

// ---------------- K3: 3x3 conv (9 shifted WMMA taps) + residual + gate -----
__global__ void __launch_bounds__(32)
k_conv3(const float* __restrict__ Wo2, const float* __restrict__ bo2,
        float* __restrict__ ws, float* __restrict__ out) {
  __shared__ float obuf[C_ * 17];
#if HAVE_TDM
  __shared__ float gqs_t[16 * C_];
  __shared__ float fcs_t[16 * C_];
#endif
  const int lane = threadIdx.x;
  const int tile = blockIdx.x;
  const int b  = tile >> 10;
  const int p0 = (tile & 1023) * 16;
  const int y  = p0 >> 7;
  const int x0 = p0 & 127;
  const int m  = lane & 15;
  const int kh = lane >> 4;

  const __bf16* midb = (const __bf16*)(ws + WS_MID_F) + (size_t)b * P_ * C_;
  const float*  gq_b = ws + WS_GQ + (size_t)b * P_ * C_;
  const float*  fc_b = ws + WS_FC + (size_t)b * P_ * C_;
  const float*  gate = ws + WS_GATE + b * C_;

  __builtin_prefetch(Wo2, 0, 1);
#if HAVE_TDM
  // prefetch the contiguous residual/conf tiles into LDS behind the WMMA work
  tdm_load_2d(gq_b + (size_t)p0 * C_, gqs_t, 16 * C_, 0, 16 * C_, 0, 16 * C_, 2);
  tdm_load_2d(fc_b + (size_t)p0 * C_, fcs_t, 16 * C_, 0, 16 * C_, 0, 16 * C_, 2);
#endif

  v8f acc[8];
#pragma unroll
  for (int nt = 0; nt < 8; ++nt) acc[nt] = splat8(bo2[nt * 16 + m]);

  for (int ty = -1; ty <= 1; ++ty) {
    int yy = y + ty;
    if (yy < 0 || yy >= 128) continue;
    for (int tx = -1; tx <= 1; ++tx) {
      int t  = (ty + 1) * 3 + (tx + 1);
      int xx = x0 + m + tx;
      bool valid = (xx >= 0 && xx < 128);
      const __bf16* arow = midb + (size_t)(yy * 128 + (valid ? xx : 0)) * C_;
#pragma unroll
      for (int ks = 0; ks < 4; ++ks) {
        v16bf a;
        if (valid) {
          int kb = ks * 32 + kh * 8;
#pragma unroll
          for (int j = 0; j < 8; ++j) { a[j] = arow[kb + j]; a[8 + j] = arow[kb + 16 + j]; }
        } else {
#pragma unroll
          for (int j = 0; j < 16; ++j) a[j] = (__bf16)0.f;
        }
#pragma unroll
        for (int nt = 0; nt < 8; ++nt) {
          int o = nt * 16 + m;
          int k = ks * 32 + kh * 8;
          v16bf bf;
#pragma unroll
          for (int j = 0; j < 16; ++j)
            bf[j] = (__bf16)Wo2[(size_t)(o * C_ + k + j + (j >> 3) * 8) * 9 + t];
          acc[nt] = wmma_bf16(a, bf, acc[nt]);
        }
      }
    }
  }

#if HAVE_TDM
  __builtin_amdgcn_s_wait_tensorcnt(0);
  __syncthreads();
#endif

  // out = gate*(conv3 + gq) + (1-gate)*fused_conf ; transpose via LDS for NCHW
#pragma unroll
  for (int nt = 0; nt < 8; ++nt) {
    int o = nt * 16 + m;
    float g = gate[o];
#pragma unroll
    for (int j = 0; j < 8; ++j) {
      int mm = j + 8 * kh;
#if HAVE_TDM
      float gqv = gqs_t[mm * C_ + o];
      float fcv = fcs_t[mm * C_ + o];
#else
      size_t pi = (size_t)(p0 + mm) * C_ + o;
      float gqv = gq_b[pi];
      float fcv = fc_b[pi];
#endif
      obuf[o * 17 + mm] = g * (acc[nt][j] + gqv) + (1.f - g) * fcv;
    }
  }
  __syncthreads();
  float* ob = out + (size_t)b * C_ * P_;
  for (int e = lane; e < C_ * 16; e += 32) {
    int o = e >> 4, px = e & 15;
    ob[(size_t)o * P_ + p0 + px] = obuf[o * 17 + px];
  }
}

extern "C" void kernel_launch(void* const* d_in, const int* in_sizes, int n_in,
                              void* d_out, int out_size, void* d_ws, size_t ws_size,
                              hipStream_t stream) {
  const float* bev = (const float*)d_in[0];
  const float* snr = (const float*)d_in[1];
  const float* vr  = (const float*)d_in[2];
  const float* Wq  = (const float*)d_in[3];
  const float* bq  = (const float*)d_in[4];
  const float* Wk  = (const float*)d_in[5];
  const float* bk  = (const float*)d_in[6];
  const float* Wv  = (const float*)d_in[7];
  const float* bv  = (const float*)d_in[8];
  const float* Wo1 = (const float*)d_in[9];
  const float* bo1 = (const float*)d_in[10];
  const float* Wo2 = (const float*)d_in[11];
  const float* bo2 = (const float*)d_in[12];
  const float* Wc1 = (const float*)d_in[13];
  const float* bc1 = (const float*)d_in[14];
  const float* Wc2 = (const float*)d_in[15];
  const float* bc2 = (const float*)d_in[16];
  const float* Wg  = (const float*)d_in[17];
  const float* bg  = (const float*)d_in[18];
  const float* Wr1 = (const float*)d_in[19];
  const float* br1 = (const float*)d_in[20];
  const float* Wr2 = (const float*)d_in[21];
  const float* br2 = (const float*)d_in[22];

  float* ws  = (float*)d_ws;
  float* out = (float*)d_out;
  float* out_attn = out + (size_t)B_ * C_ * P_;
  float* out_conf = out_attn + (size_t)B_ * N_ * P_;

  k_prep<<<3, 256, 0, stream>>>(snr, vr, Wg, bg, Wr1, br1, Wr2, br2, ws);
  k_attn<<<B_ * 1024, 32, 0, stream>>>(bev, Wq, bq, Wk, bk, Wv, bv, Wo1, bo1,
                                       Wc1, bc1, Wc2, bc2, ws, out_attn, out_conf);
  k_conv3<<<B_ * 1024, 32, 0, stream>>>(Wo2, bo2, ws, out);
}
